// HeterogeneousGNNEncoder_49976239457026
// MI455X (gfx1250) — matrix operand order repaired and checked by reading, not compiled, not run on hardware
//
#include <hip/hip_runtime.h>
#include <math.h>

#define HID 128

typedef __attribute__((ext_vector_type(16))) _Float16 v16h;
typedef __attribute__((ext_vector_type(8)))  _Float16 v8h;
typedef __attribute__((ext_vector_type(8)))  float    v8f;

// ---------------------------------------------------------------------------
// Utility
// ---------------------------------------------------------------------------
__global__ void fill_kernel(float* __restrict__ p, float v, int n) {
    int i = blockIdx.x * blockDim.x + threadIdx.x;
    if (i < n) p[i] = v;
}

__device__ __forceinline__ float leaky02(float v) { return v > 0.f ? v : 0.2f * v; }

__device__ __forceinline__ void atomicMaxF(float* addr, float val) {
    int* ai = (int*)addr;
    int old = __float_as_int(*addr);
    while (__int_as_float(old) < val) {
        int assumed = old;
        old = atomicCAS(ai, assumed, __float_as_int(val));
        if (old == assumed) break;
    }
}

// f32 -> f16 copy (node features, once per layer per type)
__global__ void cvt_f16_kernel(const float* __restrict__ in, _Float16* __restrict__ out, int n) {
    int i = blockIdx.x * blockDim.x + threadIdx.x;
    if (i < n) out[i] = (_Float16)in[i];
}

// All L*8 GAT weight matrices f32[k][n] -> f16 transposed [m][n][k] (once)
__global__ void cvt_wt_kernel(const float* __restrict__ W, _Float16* __restrict__ Wt, int nmats) {
    int i = blockIdx.x * blockDim.x + threadIdx.x;
    if (i >= nmats * 16384) return;
    int m = i >> 14, rem = i & 16383;
    int n = rem >> 7, k = rem & 127;
    Wt[i] = (_Float16)W[(size_t)m * 16384 + k * 128 + n];
}

// ---------------------------------------------------------------------------
// Input projection: out[n,128] = x[n,fin] @ W[fin,128] + b   (fin <= 10)
// ---------------------------------------------------------------------------
__global__ void proj_kernel(const float* __restrict__ x, const float* __restrict__ W,
                            const float* __restrict__ b, float* __restrict__ out,
                            int n, int fin) {
    int i = blockIdx.x * blockDim.x + threadIdx.x;
    if (i >= n * HID) return;
    int row = i >> 7, k = i & 127;
    float s = b[k];
    for (int f = 0; f < fin; ++f) s += x[(size_t)row * fin + f] * W[f * HID + k];
    out[i] = s;
}

// ---------------------------------------------------------------------------
// WMMA GEMM: Hout[n,128] = Xh[n,128] @ W[128,128]
// Inputs pre-converted to f16 (Wth already transposed: Wth[n*128+k] = W[k][n]).
// LDS tiles staged with global_load_async_to_lds_b128 (ASYNCcnt path);
// 8 wave32 per block, 64 rows/block, 16 v_wmma_f32_16x16x32_f16 per wave.
// ---------------------------------------------------------------------------
__global__ __launch_bounds__(256) void gemm_h128_wmma(
    const _Float16* __restrict__ Xh, const _Float16* __restrict__ Wth,
    float* __restrict__ Hout, int nrows) {
    __shared__ _Float16 Xs[64][136];    // padded pitch 272B (16B aligned, conflict-free)
    __shared__ _Float16 Wt[128][136];

    const int tid  = threadIdx.x;
    const int row0 = blockIdx.x * 64;

    const unsigned wt_base = (unsigned)(unsigned long long)(void*)&Wt[0][0];
    const unsigned xs_base = (unsigned)(unsigned long long)(void*)&Xs[0][0];

    // Stage W^T tile: 128 rows x 128 halves = 2048 x 16B chunks, async to LDS
#pragma unroll
    for (int it = 0; it < 8; ++it) {
        int c = tid + it * 256;
        int row = c >> 4, col8 = c & 15;
        unsigned lds = wt_base + (unsigned)(row * 272 + col8 * 16);
        int goff = c * 16;  // Wth is contiguous in the same order
        asm volatile("global_load_async_to_lds_b128 %0, %1, %2"
                     :: "v"(lds), "v"(goff), "s"(Wth) : "memory");
    }
    // Stage X rows [row0, row0+64): 1024 x 16B chunks
#pragma unroll
    for (int it = 0; it < 4; ++it) {
        int c = tid + it * 256;
        int r = c >> 4, col8 = c & 15;
        int gr = row0 + r;
        unsigned lds = xs_base + (unsigned)(r * 272 + col8 * 16);
        if (gr < nrows) {
            int goff = gr * 256 + col8 * 16;   // 128 halves = 256B per row
            asm volatile("global_load_async_to_lds_b128 %0, %1, %2"
                         :: "v"(lds), "v"(goff), "s"(Xh) : "memory");
        } else {
            *(float4*)&Xs[r][col8 * 8] = make_float4(0.f, 0.f, 0.f, 0.f);
        }
    }
    asm volatile("s_wait_asynccnt 0" ::: "memory");
    __syncthreads();

    const int wave = tid >> 5, lane = tid & 31;
    const int m16 = lane & 15, hsel = lane >> 4;
    const int mt = wave & 3, ntb = (wave >> 2) * 4;
    const int mrow = mt * 16 + m16;

    v8f zero = {0.f, 0.f, 0.f, 0.f, 0.f, 0.f, 0.f, 0.f};
    v8f acc[4];
#pragma unroll
    for (int t = 0; t < 4; ++t) acc[t] = zero;

#pragma unroll
    for (int kb = 0; kb < 128; kb += 32) {
        // A fragment (16x32 f16): lane holds K in [kb+8h,+8) u [kb+16+8h,+8)
        v8h alo = *(const v8h*)&Xs[mrow][kb + 8 * hsel];
        v8h ahi = *(const v8h*)&Xs[mrow][kb + 16 + 8 * hsel];
        v16h a;
#pragma unroll
        for (int j = 0; j < 8; ++j) { a[j] = alo[j]; a[j + 8] = ahi[j]; }
#pragma unroll
        for (int t = 0; t < 4; ++t) {
            // B fragment (32x16 f16): lane = column n, K in [kb+8h,+8) u [kb+16+8h,+8)
            int nn = (ntb + t) * 16 + m16;
            v8h blo = *(const v8h*)&Wt[nn][kb + 8 * hsel];
            v8h bhi = *(const v8h*)&Wt[nn][kb + 16 + 8 * hsel];
            v16h b;
#pragma unroll
            for (int j = 0; j < 8; ++j) { b[j] = blo[j]; b[j + 8] = bhi[j]; }
            acc[t] = __builtin_amdgcn_wmma_f32_16x16x32_f16(
                false, a, false, b, (short)0, acc[t], false, false);
        }
    }

    // C/D layout: lane = N (n=lane&15), VGPR v = row M = v + 8*(lane>>4)
    const int gm0 = row0 + mt * 16 + hsel * 8;
#pragma unroll
    for (int v = 0; v < 8; ++v) {
        int gm = gm0 + v;
        if (gm < nrows) {
#pragma unroll
            for (int t = 0; t < 4; ++t)
                Hout[(size_t)gm * HID + (ntb + t) * 16 + m16] = acc[t][v];
        }
    }
}

// ---------------------------------------------------------------------------
// Per-node attention logits: al[i,h] = sum_d H[i, h*32+d] * a[h*32+d]
// ---------------------------------------------------------------------------
__global__ void attn_kernel(const float* __restrict__ Hm, const float* __restrict__ av,
                            float* __restrict__ al, int n) {
    int i = blockIdx.x * blockDim.x + threadIdx.x;
    if (i >= n * 4) return;
    int row = i >> 2, h = i & 3;
    const float* hp = Hm + (size_t)row * HID + h * 32;
    const float* ap = av + h * 32;
    float s = 0.f;
#pragma unroll
    for (int d = 0; d < 32; ++d) s += hp[d] * ap[d];
    al[i] = s;
}

// ---------------------------------------------------------------------------
// Edge pass 1: segment max of leaky_relu(al_s[src]+al_d[dst]) per (dst, head)
// ---------------------------------------------------------------------------
__global__ void edge_max_kernel(const int* __restrict__ src, const int* __restrict__ dst,
                                const float* __restrict__ als, const float* __restrict__ ald,
                                float* __restrict__ m, int ne) {
    int e = blockIdx.x * blockDim.x + threadIdx.x;
    if (e >= ne) return;
    int s = src[e], d = dst[e];
#pragma unroll
    for (int h = 0; h < 4; ++h) {
        float v = leaky02(als[s * 4 + h] + ald[d * 4 + h]);
        atomicMaxF(&m[d * 4 + h], v);
    }
}

// ---------------------------------------------------------------------------
// Edge pass 2: softmax denominators den[dst,h] += exp(e - m[dst,h])
// ---------------------------------------------------------------------------
__global__ void edge_den_kernel(const int* __restrict__ src, const int* __restrict__ dst,
                                const float* __restrict__ als, const float* __restrict__ ald,
                                const float* __restrict__ m, float* __restrict__ den, int ne) {
    int e = blockIdx.x * blockDim.x + threadIdx.x;
    if (e >= ne) return;
    int s = src[e], d = dst[e];
#pragma unroll
    for (int h = 0; h < 4; ++h) {
        float mm = m[d * 4 + h];
        if (!(mm > -1.0e30f)) mm = 0.f;
        float v = leaky02(als[s * 4 + h] + ald[d * 4 + h]);
        atomicAdd(&den[d * 4 + h], __expf(v - mm));
    }
}

// ---------------------------------------------------------------------------
// Edge pass 3: acc[dst,k] += Hsrc[src,k] * coef -- one float4 per thread
// ---------------------------------------------------------------------------
__global__ void edge_msg_kernel(const int* __restrict__ src, const int* __restrict__ dst,
                                const float* __restrict__ als, const float* __restrict__ ald,
                                const float* __restrict__ m, const float* __restrict__ den,
                                const float* __restrict__ Hsrc, float* __restrict__ accd,
                                int ne) {
    int i = blockIdx.x * blockDim.x + threadIdx.x;
    if (i >= ne * 32) return;
    int e = i >> 5, q = i & 31;          // q: float4 index, features q*4..q*4+3
    int s = src[e], d = dst[e];
    int h = q >> 3;                      // head = (q*4)/32
    float mm = m[d * 4 + h];
    if (!(mm > -1.0e30f)) mm = 0.f;
    float v = leaky02(als[s * 4 + h] + ald[d * 4 + h]);
    float coef = __expf(v - mm) / (den[d * 4 + h] + 1e-16f);
    const float4 hv = *(const float4*)(Hsrc + (size_t)s * HID + q * 4);
    float* ap = accd + (size_t)d * HID + q * 4;
    atomicAdd(ap + 0, hv.x * coef);
    atomicAdd(ap + 1, hv.y * coef);
    atomicAdd(ap + 2, hv.z * coef);
    atomicAdd(ap + 3, hv.w * coef);
}

// ---------------------------------------------------------------------------
// HeteroConv mean + bias + relu + residual + LayerNorm, one wave32 per row
// ---------------------------------------------------------------------------
__global__ __launch_bounds__(256) void finalize_kernel(
    float* __restrict__ x, const float* __restrict__ acc,
    const float* __restrict__ b0, const float* __restrict__ b1, const float* __restrict__ b2,
    float inv_nrel, const float* __restrict__ lns, const float* __restrict__ lnb, int n) {
    int wave = threadIdx.x >> 5, lane = threadIdx.x & 31;
    int row = blockIdx.x * 8 + wave;
    if (row >= n) return;
    float vals[4];
    float s1 = 0.f;
#pragma unroll
    for (int j = 0; j < 4; ++j) {
        int k = lane * 4 + j;
        float a = acc[(size_t)row * HID + k] + b0[k];
        if (b1) a += b1[k];
        if (b2) a += b2[k];
        a *= inv_nrel;
        a = a > 0.f ? a : 0.f;                         // relu
        float v = a + x[(size_t)row * HID + k];        // residual
        vals[j] = v; s1 += v;
    }
#pragma unroll
    for (int off = 16; off > 0; off >>= 1) s1 += __shfl_xor(s1, off);
    float mu = s1 * (1.f / 128.f);
    float s2 = 0.f;
#pragma unroll
    for (int j = 0; j < 4; ++j) { float dv = vals[j] - mu; s2 += dv * dv; }
#pragma unroll
    for (int off = 16; off > 0; off >>= 1) s2 += __shfl_xor(s2, off);
    float rstd = rsqrtf(s2 * (1.f / 128.f) + 1e-5f);
#pragma unroll
    for (int j = 0; j < 4; ++j) {
        int k = lane * 4 + j;
        x[(size_t)row * HID + k] = (vals[j] - mu) * rstd * lns[k] + lnb[k];
    }
}

// ---------------------------------------------------------------------------
// Pooling: segment sums + counts, then mean written into output slab
// ---------------------------------------------------------------------------
__global__ void pool_accum_kernel(const float* __restrict__ x, const int* __restrict__ bidx,
                                  float* __restrict__ sums, float* __restrict__ cnt, int n) {
    long long i = (long long)blockIdx.x * blockDim.x + threadIdx.x;
    if (i >= (long long)n * HID) return;
    int row = (int)(i >> 7), k = (int)(i & 127);
    int g = bidx[row];
    atomicAdd(&sums[(size_t)g * HID + k], x[(size_t)row * HID + k]);
    if (k == 0) atomicAdd(&cnt[g], 1.f);
}

__global__ void pool_write_kernel(const float* __restrict__ sums, const float* __restrict__ cnt,
                                  float* __restrict__ out, int Bg) {
    int i = blockIdx.x * blockDim.x + threadIdx.x;
    int total = 4 * Bg * HID;
    if (i >= total) return;
    int t = i / (Bg * HID);
    int rem = i - t * Bg * HID;
    int g = rem >> 7, k = rem & 127;
    float c = cnt[t * Bg + g];
    if (c < 1.f) c = 1.f;
    out[(size_t)g * (4 * HID) + t * HID + k] = sums[i] / c;
}

// ---------------------------------------------------------------------------
// Host orchestration
// ---------------------------------------------------------------------------
extern "C" void kernel_launch(void* const* d_in, const int* in_sizes, int n_in,
                              void* d_out, int out_size, void* d_ws, size_t ws_size,
                              hipStream_t stream) {
    const int NB = 100000, NSp = 20000, NCr = 10000, NFa = 2000;
    const int Bg = 64, Ll = 2;

    const float* xin[4] = {(const float*)d_in[0], (const float*)d_in[1],
                           (const float*)d_in[2], (const float*)d_in[3]};
    const float* Wp[4] = {(const float*)d_in[4], (const float*)d_in[6],
                          (const float*)d_in[8], (const float*)d_in[10]};
    const float* bp[4] = {(const float*)d_in[5], (const float*)d_in[7],
                          (const float*)d_in[9], (const float*)d_in[11]};
    const int fin[4] = {8, 10, 7, 3};
    const float* gat_W  = (const float*)d_in[12];
    const float* gat_as = (const float*)d_in[13];
    const float* gat_ad = (const float*)d_in[14];
    const float* gat_b  = (const float*)d_in[15];
    const float* ln_s   = (const float*)d_in[16];
    const float* ln_b   = (const float*)d_in[17];
    const int* bidx[4] = {(const int*)d_in[34], (const int*)d_in[35],
                          (const int*)d_in[36], (const int*)d_in[37]};

    int Nt[4] = {NB, NSp, NCr, NFa};

    // Workspace carve-out (float units; all segments 16B-aligned)
    float* ws = (float*)d_ws;
    size_t o = 0;
    float* xbuf[4]; for (int t = 0; t < 4; ++t) { xbuf[t] = ws + o; o += (size_t)Nt[t] * HID; }
    float* accb[4]; for (int t = 0; t < 4; ++t) { accb[t] = ws + o; o += (size_t)Nt[t] * HID; }
    float* Hs  = ws + o; o += (size_t)NB * HID;
    float* Hd  = ws + o; o += (size_t)NB * HID;
    float* als = ws + o; o += (size_t)NB * 4;
    float* ald = ws + o; o += (size_t)NB * 4;
    float* mb  = ws + o; o += (size_t)NB * 4;
    float* db  = ws + o; o += (size_t)NB * 4;
    float* psum = ws + o; o += (size_t)4 * Bg * HID;
    float* pcnt = ws + o; o += (size_t)4 * Bg;
    _Float16* wh = (_Float16*)(ws + o); o += (size_t)16 * 16384 / 2;          // 16 W^T mats, f16
    _Float16* xh[4];
    for (int t = 0; t < 4; ++t) { xh[t] = (_Float16*)(ws + o); o += (size_t)Nt[t] * HID / 2; }

    // One-time: all GAT weights f32 -> f16 transposed
    cvt_wt_kernel<<<(16 * 16384 + 255) / 256, 256, 0, stream>>>(gat_W, wh, 16);

    // Input projections
    for (int t = 0; t < 4; ++t) {
        int work = Nt[t] * HID;
        proj_kernel<<<(work + 255) / 256, 256, 0, stream>>>(xin[t], Wp[t], bp[t],
                                                            xbuf[t], Nt[t], fin[t]);
    }

    // rel id, src type, dst type, d_in index of src edge array
    struct RelDef { int r, st, dt, si; };
    const RelDef rels[8] = {{0, 0, 1, 18}, {1, 1, 0, 20}, {2, 0, 2, 22}, {3, 2, 0, 24},
                            {4, 0, 3, 26}, {5, 0, 0, 28}, {6, 1, 3, 30}, {7, 2, 3, 32}};
    const int relset[4][3] = {{1, 3, 5}, {0, -1, -1}, {2, -1, -1}, {4, 6, 7}};
    const int nrelv[4] = {3, 1, 1, 3};

    for (int l = 0; l < Ll; ++l) {
        // f16 snapshot of current node features (used by all 8+ GEMMs this layer)
        for (int t = 0; t < 4; ++t) {
            int work = Nt[t] * HID;
            cvt_f16_kernel<<<(work + 255) / 256, 256, 0, stream>>>(xbuf[t], xh[t], work);
            fill_kernel<<<(work + 255) / 256, 256, 0, stream>>>(accb[t], 0.f, work);
        }
        for (int e = 0; e < 8; ++e) {
            const RelDef rd = rels[e];
            int ns = Nt[rd.st], nd = Nt[rd.dt];
            const _Float16* Wl = wh + (size_t)(l * 8 + rd.r) * 16384;

            gemm_h128_wmma<<<(ns + 63) / 64, 256, 0, stream>>>(xh[rd.st], Wl, Hs, ns);
            const float* Hdp = Hs;
            if (rd.dt != rd.st) {
                gemm_h128_wmma<<<(nd + 63) / 64, 256, 0, stream>>>(xh[rd.dt], Wl, Hd, nd);
                Hdp = Hd;
            }
            const float* as_v = gat_as + (size_t)(l * 8 + rd.r) * HID;
            const float* ad_v = gat_ad + (size_t)(l * 8 + rd.r) * HID;
            attn_kernel<<<(ns * 4 + 255) / 256, 256, 0, stream>>>(Hs, as_v, als, ns);
            attn_kernel<<<(nd * 4 + 255) / 256, 256, 0, stream>>>(Hdp, ad_v, ald, nd);

            fill_kernel<<<(nd * 4 + 255) / 256, 256, 0, stream>>>(mb, -__builtin_inff(), nd * 4);
            fill_kernel<<<(nd * 4 + 255) / 256, 256, 0, stream>>>(db, 0.f, nd * 4);

            const int* srcI = (const int*)d_in[rd.si];
            const int* dstI = (const int*)d_in[rd.si + 1];
            int ne = in_sizes[rd.si];
            edge_max_kernel<<<(ne + 255) / 256, 256, 0, stream>>>(srcI, dstI, als, ald, mb, ne);
            edge_den_kernel<<<(ne + 255) / 256, 256, 0, stream>>>(srcI, dstI, als, ald, mb, db, ne);
            edge_msg_kernel<<<(ne * 32 + 255) / 256, 256, 0, stream>>>(
                srcI, dstI, als, ald, mb, db, Hs, accb[rd.dt], ne);
        }
        for (int t = 0; t < 4; ++t) {
            const float* b0 = gat_b + (size_t)(l * 8 + relset[t][0]) * HID;
            const float* b1 = relset[t][1] >= 0 ? gat_b + (size_t)(l * 8 + relset[t][1]) * HID : nullptr;
            const float* b2 = relset[t][2] >= 0 ? gat_b + (size_t)(l * 8 + relset[t][2]) * HID : nullptr;
            finalize_kernel<<<(Nt[t] + 7) / 8, 256, 0, stream>>>(
                xbuf[t], accb[t], b0, b1, b2, 1.f / nrelv[t],
                ln_s + (size_t)l * HID, ln_b + (size_t)l * HID, Nt[t]);
        }
    }

    // Pooling + output
    fill_kernel<<<(4 * Bg * HID + 255) / 256, 256, 0, stream>>>(psum, 0.f, 4 * Bg * HID);
    fill_kernel<<<1, 256, 0, stream>>>(pcnt, 0.f, 4 * Bg);
    for (int t = 0; t < 4; ++t) {
        long long wm = (long long)Nt[t] * HID;
        pool_accum_kernel<<<(int)((wm + 255) / 256), 256, 0, stream>>>(
            xbuf[t], bidx[t], psum + (size_t)t * Bg * HID, pcnt + t * Bg, Nt[t]);
    }
    pool_write_kernel<<<(4 * Bg * HID + 255) / 256, 256, 0, stream>>>(
        psum, pcnt, (float*)d_out, Bg);
}
